// TripletLossInBatch_386547057031
// MI455X (gfx1250) — matrix operand order repaired and checked by reading, not compiled, not run on hardware
//
#include <hip/hip_runtime.h>
#include <hip/hip_bf16.h>

typedef _Float16 v16h __attribute__((ext_vector_type(16)));
typedef _Float16 v8h  __attribute__((ext_vector_type(8)));
typedef float    v8f  __attribute__((ext_vector_type(8)));

#define NB      4096
#define DIM     1024
#define TMARGIN 0.3f

// ---------------------------------------------------------------------------
// CDNA5 async global->LDS copy (ASYNCcnt-tracked, no VGPR round trip).
// INST_OFFSET is added to BOTH the LDS and global addresses, so one base
// (lds_addr VGPR, 64-bit global addr VGPR pair) covers both 16B halves.
// ---------------------------------------------------------------------------
__device__ __forceinline__ void async_stage_32B(unsigned lds_addr,
                                                unsigned long long gaddr)
{
    asm volatile(
        "global_load_async_to_lds_b128 %0, %1, off\n\t"
        "global_load_async_to_lds_b128 %0, %1, off offset:16"
        :: "v"(lds_addr), "v"(gaddr) : "memory");
}

__device__ __forceinline__ void wait_asynccnt0()
{
    asm volatile("s_wait_asynccnt 0x0" ::: "memory");
}

// ---------------------------------------------------------------------------
// Kernel 1: row-wise normalization of both matrices + pos_dist.
// ---------------------------------------------------------------------------
__global__ __launch_bounds__(256) void nrm_kernel(
    const float* __restrict__ za, const float* __restrict__ zp,
    _Float16* __restrict__ a16, _Float16* __restrict__ p16,
    float* __restrict__ pos_dist)
{
    const int row = blockIdx.x;
    const int t   = threadIdx.x;
    const float* ar = za + (size_t)row * DIM;
    const float* pr = zp + (size_t)row * DIM;

    float av[4], pv[4];
    float aa = 0.f, pp = 0.f, ap = 0.f;
#pragma unroll
    for (int i = 0; i < 4; ++i) {
        const int e = t + i * 256;
        av[i] = ar[e]; pv[i] = pr[e];
        aa += av[i] * av[i];
        pp += pv[i] * pv[i];
        ap += av[i] * pv[i];
    }
#pragma unroll
    for (int m = 16; m >= 1; m >>= 1) {
        aa += __shfl_xor(aa, m, 32);
        pp += __shfl_xor(pp, m, 32);
        ap += __shfl_xor(ap, m, 32);
    }
    __shared__ float s[3][8];
    const int wave = t >> 5, lane = t & 31;
    if (lane == 0) { s[0][wave] = aa; s[1][wave] = pp; s[2][wave] = ap; }
    __syncthreads();
    if (t == 0) {
        float A = 0.f, P = 0.f, Q = 0.f;
        for (int w = 0; w < 8; ++w) { A += s[0][w]; P += s[1][w]; Q += s[2][w]; }
        const float ia = rsqrtf(A), ip = rsqrtf(P);
        s[0][0] = ia; s[1][0] = ip;
        pos_dist[row] = 1.0f - Q * ia * ip;
    }
    __syncthreads();
    const float ia = s[0][0], ip = s[1][0];
    _Float16* aout = a16 + (size_t)row * DIM;
    _Float16* pout = p16 + (size_t)row * DIM;
#pragma unroll
    for (int i = 0; i < 4; ++i) {
        const int e = t + i * 256;
        aout[e] = (_Float16)(av[i] * ia);
        pout[e] = (_Float16)(pv[i] * ip);
    }
}

// ---------------------------------------------------------------------------
// Kernel 2: initialize per-row mining stats + loss accumulator.
// ---------------------------------------------------------------------------
__global__ __launch_bounds__(256) void init_kernel(
    float* __restrict__ cnt, float* __restrict__ sum,
    unsigned* __restrict__ mnu, float* __restrict__ out)
{
    const int i = blockIdx.x * 256 + threadIdx.x;
    if (i < NB) { cnt[i] = 0.f; sum[i] = 0.f; mnu[i] = 0xFFFFFFFFu; }
    if (i == 0) out[0] = 0.f;
}

// ---------------------------------------------------------------------------
// Kernel 3: fused 128x128-tile WMMA GEMM + semi-hard-mining epilogue.
// - B tile double-buffered in LDS, staged with global_load_async_to_lds_b128
//   (issue next chunk's copy before the current chunk's WMMAs; one
//   s_wait_asynccnt + barrier per K-step).
// - A fragments software-pipelined one K-step ahead in registers.
// - B fragments double-buffered in REGISTERS across the 8 column tiles, so
//   the compiler can issue tile tt+1's ds_loads before WMMA tt and use a
//   partial s_wait_dscnt instead of a full DS drain per WMMA.
// ---------------------------------------------------------------------------
__global__ __launch_bounds__(256) void gemm_stats_kernel(
    const _Float16* __restrict__ a16, const _Float16* __restrict__ p16,
    const float* __restrict__ pos_dist,
    float* __restrict__ cnt, float* __restrict__ sum,
    unsigned* __restrict__ mnu)
{
    __shared__ _Float16 Bs[2][128 * 32];   // double-buffered [col][k], 2 x 8 KB

    const int t    = threadIdx.x;
    const int wave = t >> 5;
    const int ln   = t & 31;
    const int half = (ln < 16) ? 0 : 1;    // lane half selects K sub-range
    const int lr   = ln & 15;

    const int mBase = blockIdx.y * 128 + wave * 16;
    const int nBase = blockIdx.x * 128;

    // pos_dist for this lane's 8 output rows (C layout: M = r + half*8)
    float pdv[8];
#pragma unroll
    for (int r = 0; r < 8; ++r) pdv[r] = pos_dist[mBase + r + half * 8];

    v8f acc[8] = {};

    // A fragment: row = mBase + (ln&15); K-halves 0-15 vs 16-31
    const _Float16* arow = a16 + (size_t)(mBase + lr) * DIM;
    const int aoff = half * 16;

    // async staging: thread -> (col, 16-K half), 32 B per thread per step
    const int scol  = t >> 1;
    const int shalf = t & 1;
    const _Float16* sptr = p16 + (size_t)(nBase + scol) * DIM + shalf * 16;
    const unsigned ldsA =
        (unsigned)(unsigned long long)(void*)&Bs[0][scol * 32 + shalf * 16];
    const unsigned ldsB =
        (unsigned)(unsigned long long)(void*)&Bs[1][scol * 32 + shalf * 16];

    // prologue: stage k=0 into buffer 0, preload A fragment for k=0
    async_stage_32B(ldsA, (unsigned long long)(sptr));
    union { v16h v; v8h h[2]; } afCur;
    afCur.h[0] = *(const v8h*)(arow + aoff);
    afCur.h[1] = *(const v8h*)(arow + aoff + 8);
    wait_asynccnt0();
    __syncthreads();

    int cur = 0;
    for (int k0 = 0; k0 < DIM; k0 += 32, cur ^= 1) {
        const int kn = k0 + 32;
        const int kk = (kn < DIM) ? kn : 0;      // wrapped: branchless prefetch

        // 1) issue async stage of NEXT chunk into the other buffer
        async_stage_32B(cur ? ldsA : ldsB, (unsigned long long)(sptr + kk));

        // 2) prefetch next A fragment (global latency overlaps WMMAs)
        const v8h an0 = *(const v8h*)(arow + kk + aoff);
        const v8h an1 = *(const v8h*)(arow + kk + aoff + 8);

        // 3) 8 WMMAs, B fragments double-buffered in registers
        const _Float16* bwave = &Bs[cur][lr * 32 + half * 16];
        union { v16h v; v8h h[2]; } bf0, bf1;
        bf0.h[0] = *(const v8h*)(bwave);
        bf0.h[1] = *(const v8h*)(bwave + 8);
#pragma unroll
        for (int tt = 0; tt < 8; ++tt) {
            const int nt = (tt < 7) ? tt + 1 : 7;      // next tile (7: dead)
            const _Float16* bp = bwave + nt * 16 * 32;
            if ((tt & 1) == 0) {
                bf1.h[0] = *(const v8h*)(bp);
                bf1.h[1] = *(const v8h*)(bp + 8);
                acc[tt] = __builtin_amdgcn_wmma_f32_16x16x32_f16(
                    false, afCur.v, false, bf0.v, (short)0, acc[tt], false, false);
            } else {
                bf0.h[0] = *(const v8h*)(bp);
                bf0.h[1] = *(const v8h*)(bp + 8);
                acc[tt] = __builtin_amdgcn_wmma_f32_16x16x32_f16(
                    false, afCur.v, false, bf1.v, (short)0, acc[tt], false, false);
            }
        }

        // 4) next chunk resident + all waves done reading current buffer
        wait_asynccnt0();
        __syncthreads();
        afCur.h[0] = an0;
        afCur.h[1] = an1;
    }

    // ---- mining epilogue: per-lane partial stats over this block's columns
    float c8[8], s8[8], m8[8];
#pragma unroll
    for (int r = 0; r < 8; ++r) { c8[r] = 0.f; s8[r] = 0.f; m8[r] = 3.0e38f; }

#pragma unroll
    for (int tt = 0; tt < 8; ++tt) {
        const int col = nBase + tt * 16 + lr;
#pragma unroll
        for (int r = 0; r < 8; ++r) {
            const int row = mBase + r + half * 8;
            const float nd = 1.0f - acc[tt][r];
            if (col != row) {                                  // off-diagonal
                const float pd = pdv[r];
                const bool sh = (nd > pd) && (nd < pd + TMARGIN);
                if (sh) { c8[r] += 1.f; s8[r] += nd; }
                m8[r] = fminf(m8[r], nd);
            }
        }
    }

    // reduce across the 16 lanes that share each output row
#pragma unroll
    for (int m = 8; m >= 1; m >>= 1) {
#pragma unroll
        for (int r = 0; r < 8; ++r) {
            c8[r] += __shfl_xor(c8[r], m, 16);
            s8[r] += __shfl_xor(s8[r], m, 16);
            m8[r]  = fminf(m8[r], __shfl_xor(m8[r], m, 16));
        }
    }

    if (lr == 0) {
#pragma unroll
        for (int r = 0; r < 8; ++r) {
            const int row = mBase + r + half * 8;
            if (c8[r] > 0.f) {
                atomicAdd(&cnt[row], c8[r]);
                atomicAdd(&sum[row], s8[r]);
            }
            // float -> order-preserving uint (handles sign) for atomic min
            const unsigned b = __float_as_uint(m8[r]);
            const unsigned u = (b & 0x80000000u) ? ~b : (b | 0x80000000u);
            atomicMin(&mnu[row], u);
        }
    }
}

// ---------------------------------------------------------------------------
// Kernel 4: per-row neg_dist selection + relu + mean into d_out.
// ---------------------------------------------------------------------------
__global__ __launch_bounds__(256) void finalize_kernel(
    const float* __restrict__ pos_dist, const float* __restrict__ cnt,
    const float* __restrict__ sum, const unsigned* __restrict__ mnu,
    float* __restrict__ out)
{
    const int row = blockIdx.x * 256 + threadIdx.x;
    const float c = cnt[row];
    float neg;
    if (c > 0.f) {
        neg = sum[row] / c;
    } else {
        const unsigned u = mnu[row];
        const unsigned b = (u & 0x80000000u) ? (u & 0x7FFFFFFFu) : ~u;
        neg = __uint_as_float(b);
    }
    float l = fmaxf(0.f, pos_dist[row] - neg + TMARGIN);
#pragma unroll
    for (int m = 16; m >= 1; m >>= 1) l += __shfl_xor(l, m, 32);
    __shared__ float ws[8];
    const int wave = threadIdx.x >> 5, lane = threadIdx.x & 31;
    if (lane == 0) ws[wave] = l;
    __syncthreads();
    if (threadIdx.x == 0) {
        float ttl = 0.f;
        for (int w = 0; w < 8; ++w) ttl += ws[w];
        atomicAdd(out, ttl / (float)NB);
    }
}

// ---------------------------------------------------------------------------
extern "C" void kernel_launch(void* const* d_in, const int* in_sizes, int n_in,
                              void* d_out, int out_size, void* d_ws, size_t ws_size,
                              hipStream_t stream)
{
    (void)in_sizes; (void)n_in; (void)out_size; (void)ws_size;
    const float* za = (const float*)d_in[0];
    const float* zp = (const float*)d_in[1];
    float* out = (float*)d_out;

    char* ws = (char*)d_ws;
    const size_t szMat = (size_t)NB * DIM * sizeof(_Float16);   // 8 MB
    _Float16* a16 = (_Float16*)(ws);
    _Float16* p16 = (_Float16*)(ws + szMat);
    float*    pd  = (float*)   (ws + 2 * szMat);
    float*    cnt = (float*)   (ws + 2 * szMat + 1 * NB * sizeof(float));
    float*    sum = (float*)   (ws + 2 * szMat + 2 * NB * sizeof(float));
    unsigned* mnu = (unsigned*)(ws + 2 * szMat + 3 * NB * sizeof(float));

    nrm_kernel<<<NB, 256, 0, stream>>>(za, zp, a16, p16, pd);
    init_kernel<<<NB / 256, 256, 0, stream>>>(cnt, sum, mnu, out);
    dim3 grid(NB / 128, NB / 128);
    gemm_stats_kernel<<<grid, 256, 0, stream>>>(a16, p16, pd, cnt, sum, mnu);
    finalize_kernel<<<NB / 256, 256, 0, stream>>>(pd, cnt, sum, mnu, out);
}
